// Net_61555471286807
// MI455X (gfx1250) — matrix operand order repaired and checked by reading, not compiled, not run on hardware
//
#include <hip/hip_runtime.h>
#include <stdint.h>

typedef __attribute__((ext_vector_type(16))) __bf16 bf16x16;
typedef __attribute__((ext_vector_type(8)))  float  floatx8;

union FragAB { bf16x16 v; uint4 q[2]; };

__device__ __forceinline__ int imin(int a, int b) { return a < b ? a : b; }

// ---------------------------------------------------------------------------
// elementwise f32 -> bf16 convert
// ---------------------------------------------------------------------------
__global__ void f32_to_bf16_kernel(const float* __restrict__ in,
                                   __bf16* __restrict__ out, int n) {
  int i = blockIdx.x * blockDim.x + threadIdx.x;
  if (i < n) out[i] = (__bf16)in[i];
}

// ---------------------------------------------------------------------------
// build X0^T:  x (B=64, C=8, V=4096)  ->  Xt[(c*64+b)*4096 + v]  (f32 + bf16)
// ---------------------------------------------------------------------------
__global__ void build_x0_kernel(const float* __restrict__ x,
                                float* __restrict__ Xf,
                                __bf16* __restrict__ Xb) {
  int o = blockIdx.x * blockDim.x + threadIdx.x;   // 512*4096 total
  int v = o & 4095;
  int j = o >> 12;          // c*64 + b
  int c = j >> 6;
  int b = j & 63;
  float val = x[((size_t)(b * 8 + c)) * 4096 + v];
  Xf[o] = val;
  Xb[o] = (__bf16)val;
}

// ---------------------------------------------------------------------------
// bf16 WMMA GEMM (NT):  C = alpha * A @ B + beta * Prev   (+bias, +relu)
//   A  : M x K row-major bf16
//   Bt : N x K row-major bf16 (i.e. B transposed)
//   Output stored TRANSPOSED: C_t[n*M + m]  (f32 and/or bf16)
// Each wave computes a 64x64 tile (4x4 grid of 16x16x32 WMMAs).
// Workgroup = 4 waves (128 threads) covering a 128x128 block.
// ---------------------------------------------------------------------------
__global__ __launch_bounds__(128)
void gemm_bf16_nt(const __bf16* __restrict__ A, int lda,
                  const __bf16* __restrict__ Bt, int ldb,
                  float* __restrict__ Cf, __bf16* __restrict__ Cb,
                  const float* __restrict__ Prev,
                  const float* __restrict__ bias,
                  int M, int N, int K, float alpha, float beta, int relu) {
  const int lane = threadIdx.x & 31;
  const int wave = threadIdx.x >> 5;
  const int m0 = blockIdx.x * 128 + (wave >> 1) * 64;
  const int n0 = blockIdx.y * 128 + (wave & 1) * 64;
  const int lm = lane & 15;
  const int h  = lane >> 4;

  floatx8 acc[4][4] = {};   // 16 accumulators, 8 VGPRs each

  // A fragment rows (lanes 0-15 and 16-31 both map to rows M=0..15 of tile)
  int arow[4], bcol[4];
#pragma unroll
  for (int i = 0; i < 4; i++) arow[i] = imin(m0 + i * 16 + lm, M - 1);
#pragma unroll
  for (int j = 0; j < 4; j++) bcol[j] = imin(n0 + j * 16 + lm, N - 1);
  const int akoff = h * 8;    // A: half-lanes split K 0..15 / 16..31 as {0..7},{16..23} vs {8..15},{24..31}
  const int bkoff = h * 8;    // B: same K split as A: {0..7},{16..23} vs {8..15},{24..31}

  for (int kk = 0; kk < K; kk += 32) {
    FragAB a[4], b[4];
#pragma unroll
    for (int i = 0; i < 4; i++) {
      const __bf16* p = A + (size_t)arow[i] * lda + kk + akoff;
      a[i].q[0] = *(const uint4*)(p);        // VGPR0-3 : K = kk + 8h + 0..7
      a[i].q[1] = *(const uint4*)(p + 16);   // VGPR4-7 : K = kk + 16 + 8h + 0..7
    }
#pragma unroll
    for (int j = 0; j < 4; j++) {
      const __bf16* p = Bt + (size_t)bcol[j] * ldb + kk + bkoff;
      b[j].q[0] = *(const uint4*)(p);        // K = kk + 8h + 0..7
      b[j].q[1] = *(const uint4*)(p + 16);   // K = kk + 16 + 8h + 0..7
    }
#pragma unroll
    for (int i = 0; i < 4; i++)
#pragma unroll
      for (int j = 0; j < 4; j++)
        acc[i][j] = __builtin_amdgcn_wmma_f32_16x16x32_bf16(
            false, a[i].v, false, b[j].v, (short)0, acc[i][j], false, false);
  }

  // Epilogue: C/D 16x16 f32 layout -> lane gives (n = lane&15, m = r + 8*(lane>>4))
#pragma unroll
  for (int j = 0; j < 4; j++) {
    int n = n0 + j * 16 + lm;
    if (n >= N) continue;
    float bv = bias ? bias[n] : 0.f;
#pragma unroll
    for (int i = 0; i < 4; i++) {
#pragma unroll
      for (int r = 0; r < 8; r++) {
        int m = m0 + i * 16 + r + 8 * h;
        if (m >= M) continue;
        size_t idx = (size_t)n * M + m;    // transposed output, ldc = M
        float y = alpha * acc[i][j][r];
        if (beta != 0.f) y += beta * Prev[idx];
        y += bv;
        if (relu) y = y > 0.f ? y : 0.f;
        if (Cf) Cf[idx] = y;
        if (Cb) Cb[idx] = (__bf16)y;
      }
    }
  }
}

// ---------------------------------------------------------------------------
// Stage-1 weight projection, accumulated per Chebyshev order k:
//   out1[b][f][v] (+)= sum_c Xk_t[(c*64+b)*4096+v] * W1[f][c*25+k]
// ---------------------------------------------------------------------------
__global__ void proj1_kernel(const float* __restrict__ Xk,
                             const float* __restrict__ W1,
                             float* __restrict__ out1, int k, int init) {
  int t = blockIdx.x * blockDim.x + threadIdx.x;  // 64*4096
  int v = t & 4095;
  int b = t >> 12;
  float xv[8];
#pragma unroll
  for (int c = 0; c < 8; c++) xv[c] = Xk[(size_t)(c * 64 + b) * 4096 + v];
#pragma unroll 4
  for (int f = 0; f < 32; f++) {
    float s = 0.f;
#pragma unroll
    for (int c = 0; c < 8; c++) s += xv[c] * W1[f * 200 + c * 25 + k];
    size_t o = ((size_t)b * 32 + f) * 4096 + v;
    out1[o] = init ? s : (out1[o] + s);
  }
}

// Stage-2 projection: out2[b][f2][v2] (+)= sum_c2 X2_t[(c2*64+b)*1024+v2]*W2[f2][c2*25+k]
__global__ void proj2_kernel(const float* __restrict__ Xk,
                             const float* __restrict__ W2,
                             float* __restrict__ out2, int k, int init) {
  int t = blockIdx.x * blockDim.x + threadIdx.x;  // 64*1024
  int v = t & 1023;
  int b = t >> 10;
  float xv[32];
#pragma unroll
  for (int c = 0; c < 32; c++) xv[c] = Xk[(size_t)(c * 64 + b) * 1024 + v];
  for (int f = 0; f < 64; f++) {
    float s = 0.f;
#pragma unroll
    for (int c = 0; c < 32; c++) s += xv[c] * W2[f * 800 + c * 25 + k];
    size_t o = ((size_t)b * 64 + f) * 1024 + v;
    out2[o] = init ? s : (out2[o] + s);
  }
}

// ---------------------------------------------------------------------------
// relu(x + bias) then maxpool4 over V; write stage-2 X0 in transposed layout
// out1 (B=64,F=32,V=4096) -> Xt[(f*64+b)*1024 + v2]  (f32 + bf16)
// ---------------------------------------------------------------------------
__global__ void pool1_kernel(const float* __restrict__ out1,
                             const float* __restrict__ b1,
                             float* __restrict__ Xf, __bf16* __restrict__ Xb) {
  int t = blockIdx.x * blockDim.x + threadIdx.x;  // 64*32*1024
  int v2 = t & 1023;
  int f  = (t >> 10) & 31;
  int b  = t >> 15;
  const float* p = out1 + ((size_t)b * 32 + f) * 4096 + v2 * 4;
  float m = fmaxf(fmaxf(p[0], p[1]), fmaxf(p[2], p[3]));
  m = fmaxf(m + b1[f], 0.f);
  size_t o = (size_t)(f * 64 + b) * 1024 + v2;
  Xf[o] = m;
  Xb[o] = (__bf16)m;
}

// out2 (B=64,F=64,V=1024) -> act bf16 [b][f*256+u]  (flatten order of reference)
__global__ void pool2_kernel(const float* __restrict__ out2,
                             const float* __restrict__ b2,
                             __bf16* __restrict__ act) {
  int t = blockIdx.x * blockDim.x + threadIdx.x;  // 64*64*256
  int u = t & 255;
  int f = (t >> 8) & 63;
  int b = t >> 14;
  const float* p = out2 + ((size_t)b * 64 + f) * 1024 + u * 4;
  float m = fmaxf(fmaxf(p[0], p[1]), fmaxf(p[2], p[3]));
  m = fmaxf(m + b2[f], 0.f);
  act[(size_t)b * 16384 + f * 256 + u] = (__bf16)m;
}

// ---------------------------------------------------------------------------
// fc2 (tiny): out[b][j] = fc2_b[j] + sum_n fc1t[n*64+b] * fc2_w[j*512+n]
// fc1t already has bias+relu applied by the GEMM epilogue.
// ---------------------------------------------------------------------------
__global__ void fc2_kernel(const float* __restrict__ fc1t,
                           const float* __restrict__ w,
                           const float* __restrict__ bias,
                           float* __restrict__ out) {
  int t = blockIdx.x * blockDim.x + threadIdx.x;
  if (t >= 640) return;
  int j = t % 10;
  int b = t / 10;
  float s = bias[j];
  for (int n = 0; n < 512; n++) s += fc1t[(size_t)n * 64 + b] * w[j * 512 + n];
  out[b * 10 + j] = s;
}

// ---------------------------------------------------------------------------
extern "C" void kernel_launch(void* const* d_in, const int* in_sizes, int n_in,
                              void* d_out, int out_size, void* d_ws, size_t ws_size,
                              hipStream_t stream) {
  (void)in_sizes; (void)n_in; (void)out_size; (void)ws_size;
  const float* x     = (const float*)d_in[0];
  const float* lap1  = (const float*)d_in[1];
  const float* lap2  = (const float*)d_in[2];
  const float* W1    = (const float*)d_in[3];
  const float* b1    = (const float*)d_in[4];
  const float* W2    = (const float*)d_in[5];
  const float* b2    = (const float*)d_in[6];
  const float* fc1w  = (const float*)d_in[7];
  const float* fc1b  = (const float*)d_in[8];
  const float* fc2w  = (const float*)d_in[9];
  const float* fc2b  = (const float*)d_in[10];
  float* out = (float*)d_out;

  // -------- workspace layout (256B aligned) --------
  char* ws = (char*)d_ws;
  size_t off = 0;
  auto alloc = [&](size_t bytes) -> char* {
    char* p = ws + off;
    off += (bytes + 255) & ~(size_t)255;
    return p;
  };
  __bf16* lap1b = (__bf16*)alloc(4096ull * 4096 * 2);
  __bf16* lap2b = (__bf16*)alloc(1024ull * 1024 * 2);
  __bf16* fc1wb = (__bf16*)alloc(512ull * 16384 * 2);
  float*  Xf[3];
  for (int i = 0; i < 3; i++) Xf[i] = (float*)alloc(2097152ull * 4);   // max(512*4096, 2048*1024)
  __bf16* Xb[2];
  for (int i = 0; i < 2; i++) Xb[i] = (__bf16*)alloc(2097152ull * 2);
  float* out1 = (float*)alloc(64ull * 32 * 4096 * 4);
  float* out2 = (float*)alloc(64ull * 64 * 1024 * 4);
  __bf16* act = (__bf16*)alloc(64ull * 16384 * 2);
  float* fc1o = (float*)alloc(512ull * 64 * 4);

  // -------- one-time bf16 conversions (re-done every call: deterministic) ---
  f32_to_bf16_kernel<<<(4096 * 4096) / 256, 256, 0, stream>>>(lap1, lap1b, 4096 * 4096);
  f32_to_bf16_kernel<<<(1024 * 1024) / 256, 256, 0, stream>>>(lap2, lap2b, 1024 * 1024);
  f32_to_bf16_kernel<<<(512 * 16384) / 256, 256, 0, stream>>>(fc1w, fc1wb, 512 * 16384);

  // -------- stage 1: Chebyshev K=25 on V=4096, N=512 columns --------
  build_x0_kernel<<<(512 * 4096) / 256, 256, 0, stream>>>(x, Xf[0], Xb[0]);
  proj1_kernel<<<(64 * 4096) / 256, 256, 0, stream>>>(Xf[0], W1, out1, 0, 1);
  for (int k = 1; k < 25; k++) {
    float alpha = (k == 1) ? 1.f : 2.f;
    float beta  = (k == 1) ? 0.f : -1.f;
    dim3 g(4096 / 128, 512 / 128);
    gemm_bf16_nt<<<g, 128, 0, stream>>>(lap1b, 4096, Xb[(k - 1) & 1], 4096,
                                        Xf[k % 3], Xb[k & 1], Xf[(k + 1) % 3],
                                        nullptr, 4096, 512, 4096, alpha, beta, 0);
    proj1_kernel<<<(64 * 4096) / 256, 256, 0, stream>>>(Xf[k % 3], W1, out1, k, 0);
  }

  // -------- relu + pool4 -> stage-2 X0 (transposed layout) --------
  pool1_kernel<<<(64 * 32 * 1024) / 256, 256, 0, stream>>>(out1, b1, Xf[0], Xb[0]);

  // -------- stage 2: Chebyshev K=25 on V=1024, N=2048 columns --------
  proj2_kernel<<<(64 * 1024) / 256, 256, 0, stream>>>(Xf[0], W2, out2, 0, 1);
  for (int k = 1; k < 25; k++) {
    float alpha = (k == 1) ? 1.f : 2.f;
    float beta  = (k == 1) ? 0.f : -1.f;
    dim3 g(1024 / 128, 2048 / 128);
    gemm_bf16_nt<<<g, 128, 0, stream>>>(lap2b, 1024, Xb[(k - 1) & 1], 1024,
                                        Xf[k % 3], Xb[k & 1], Xf[(k + 1) % 3],
                                        nullptr, 1024, 2048, 1024, alpha, beta, 0);
    proj2_kernel<<<(64 * 1024) / 256, 256, 0, stream>>>(Xf[k % 3], W2, out2, k, 0);
  }

  // -------- relu + pool4 -> flattened activations (bf16) --------
  pool2_kernel<<<(64 * 64 * 256) / 256, 256, 0, stream>>>(out2, b2, act);

  // -------- fc1 via WMMA GEMM: (64 x 16384) @ (16384 x 512), bias+relu ------
  {
    dim3 g(1, 512 / 128);
    gemm_bf16_nt<<<g, 128, 0, stream>>>(act, 16384, fc1wb, 16384,
                                        fc1o, nullptr, nullptr, fc1b,
                                        64, 512, 16384, 1.f, 0.f, 1);
  }

  // -------- fc2 (tiny VALU kernel) --------
  fc2_kernel<<<3, 256, 0, stream>>>(fc1o, fc2w, fc2b, out);
}
